// AtrousMultiHeadedAttention_11441792876630
// MI455X (gfx1250) — matrix operand order repaired and checked by reading, not compiled, not run on hardware
//
#include <hip/hip_runtime.h>
#include <hip/hip_bf16.h>
#include <cstdint>

// ---------------------------------------------------------------------------
// Atrous multi-head attention for MI455X (gfx1250), wave32 + WMMA bf16.
// Pipeline: cvt-to-bf16 -> QKV WMMA GEMMs -> flash attention (WMMA) ->
//           output-projection WMMA GEMM (f32) -> residual + LayerNorm.
// GEMM/attention use double-buffered LDS tiles; global->LDS moves use the
// CDNA5 async-to-LDS path (ASYNCcnt) when the toolchain exposes the builtins.
// ---------------------------------------------------------------------------

typedef __attribute__((ext_vector_type(16))) __bf16 v16bf;
typedef __attribute__((ext_vector_type(8)))  float  v8f;

#if __has_builtin(__builtin_amdgcn_global_load_async_to_lds_b128) && \
    __has_builtin(__builtin_amdgcn_s_wait_asynccnt)
#define HAVE_ASYNC_LDS 1
#else
#define HAVE_ASYNC_LDS 0
#endif

#if HAVE_ASYNC_LDS
// Builtin prototype (from hipcc diagnostics): b128 variant takes pointers to a
// 16-byte int vector in addrspace(1) (global) and addrspace(3) (LDS).
typedef int v4i_ls __attribute__((vector_size(16)));
typedef __attribute__((address_space(1))) v4i_ls as1_v4i;
typedef __attribute__((address_space(3))) v4i_ls as3_v4i;
// Generic LDS pointer: low 32 bits are the DS byte address -> addrspace(3).
__device__ inline as3_v4i* to_as3(void* p) {
    return (as3_v4i*)(uint32_t)(uintptr_t)p;
}
__device__ inline as1_v4i* to_as1(const void* p) {
    return (as1_v4i*)(uintptr_t)p;
}
#endif

union FragU { uint4 q[2]; v16bf v; __bf16 e[16]; };

__device__ inline v16bf load_frag2x16B(const __bf16* p0, const __bf16* p1) {
    FragU u;
    u.q[0] = *reinterpret_cast<const uint4*>(p0);
    u.q[1] = *reinterpret_cast<const uint4*>(p1);
    return u.v;
}

__device__ inline v8f zero8() {
    v8f z;
#pragma unroll
    for (int i = 0; i < 8; ++i) z[i] = 0.f;
    return z;
}

// ---------------------------------------------------------------------------
// f32 -> bf16 conversion (vectorized by 4)
// ---------------------------------------------------------------------------
__global__ __launch_bounds__(256) void cvt_f32_bf16(const float* __restrict__ in,
                                                    __bf16* __restrict__ out, int n4) {
    int i = blockIdx.x * 256 + threadIdx.x;
    if (i < n4) {
        float4 f = reinterpret_cast<const float4*>(in)[i];
        union { __bf16 o[4]; uint2 u; } pk;
        pk.o[0] = (__bf16)f.x; pk.o[1] = (__bf16)f.y;
        pk.o[2] = (__bf16)f.z; pk.o[3] = (__bf16)f.w;
        reinterpret_cast<uint2*>(out)[i] = pk.u;
    }
}

// ---------------------------------------------------------------------------
// Tiled WMMA bf16 GEMM:  C[M][N] = A[M][K] * W[N][K]^T + bias
// BM=BN=128, BK=64; 256 threads = 8 waves; wave -> 64x32 (4x2 wmma accs).
// Double-buffered LDS, async global->LDS when available.
// outb (bf16) or outf (f32) selects the output path.
// ---------------------------------------------------------------------------
__global__ __launch_bounds__(256) void gemm_bf16_wmma(
    const __bf16* __restrict__ A, const __bf16* __restrict__ W,
    const float* __restrict__ bias,
    __bf16* __restrict__ outb, float* __restrict__ outf,
    int Mdim, int Ndim, int Kdim) {
    constexpr int BK = 64, PAD = 8, LDST = BK + PAD;   // 144B stride, 16B aligned rows
    __shared__ __bf16 As[2][128][LDST];
    __shared__ __bf16 Bs[2][128][LDST];

    const int tid = threadIdx.x;
    const int wave = tid >> 5, lane = tid & 31;
    const int lhalf = lane >> 4, l15 = lane & 15;
    const int wm = wave >> 2, wn = wave & 3;           // 2 x 4 wave grid
    const int m0 = blockIdx.x * 128, n0 = blockIdx.y * 128;
    const int lrow = tid >> 1, lcol = (tid & 1) * 32;  // 64B per thread per tile

    const __bf16* aBase = A + (size_t)(m0 + lrow) * Kdim + lcol;
    const __bf16* wBase = W + (size_t)(n0 + lrow) * Kdim + lcol;

    v8f acc[4][2];
#pragma unroll
    for (int mi = 0; mi < 4; ++mi)
#pragma unroll
        for (int nj = 0; nj < 2; ++nj) acc[mi][nj] = zero8();

    auto issue_tile = [&](int kt, int buf) {
        const __bf16* ap = aBase + kt * BK;
        const __bf16* wp = wBase + kt * BK;
#if HAVE_ASYNC_LDS
#pragma unroll
        for (int i = 0; i < 4; ++i) {
            __builtin_amdgcn_global_load_async_to_lds_b128(
                to_as1(ap + 8 * i), to_as3(&As[buf][lrow][lcol + 8 * i]), 0, 0);
            __builtin_amdgcn_global_load_async_to_lds_b128(
                to_as1(wp + 8 * i), to_as3(&Bs[buf][lrow][lcol + 8 * i]), 0, 0);
        }
#else
        uint4 ar[4], wr[4];
#pragma unroll
        for (int i = 0; i < 4; ++i) {
            ar[i] = *reinterpret_cast<const uint4*>(ap + 8 * i);
            wr[i] = *reinterpret_cast<const uint4*>(wp + 8 * i);
        }
#pragma unroll
        for (int i = 0; i < 4; ++i) {
            *reinterpret_cast<uint4*>(&As[buf][lrow][lcol + 8 * i]) = ar[i];
            *reinterpret_cast<uint4*>(&Bs[buf][lrow][lcol + 8 * i]) = wr[i];
        }
#endif
    };

    const int KT = Kdim / BK;
    issue_tile(0, 0);
    for (int kt = 0; kt < KT; ++kt) {
        const int buf = kt & 1;
        if (kt + 1 < KT) {
            issue_tile(kt + 1, buf ^ 1);
#if HAVE_ASYNC_LDS
            __builtin_amdgcn_s_wait_asynccnt(8);  // newest 8 = next tile; current done
#endif
        } else {
#if HAVE_ASYNC_LDS
            __builtin_amdgcn_s_wait_asynccnt(0);
#endif
        }
        __syncthreads();

#pragma unroll
        for (int kc = 0; kc < 2; ++kc) {
            const int ks = kc * 32 + lhalf * 8;    // A frag K offset (ISA layout)
            const int kb = kc * 32 + lhalf * 8;    // B frag K offset (ISA layout)
            v16bf af[4], bfg[2];
#pragma unroll
            for (int mi = 0; mi < 4; ++mi) {
                const int r = wm * 64 + mi * 16 + l15;
                af[mi] = load_frag2x16B(&As[buf][r][ks], &As[buf][r][ks + 16]);
            }
#pragma unroll
            for (int nj = 0; nj < 2; ++nj) {
                const int r = wn * 32 + nj * 16 + l15;
                bfg[nj] = load_frag2x16B(&Bs[buf][r][kb], &Bs[buf][r][kb + 16]);
            }
#pragma unroll
            for (int mi = 0; mi < 4; ++mi)
#pragma unroll
                for (int nj = 0; nj < 2; ++nj)
                    acc[mi][nj] = __builtin_amdgcn_wmma_f32_16x16x32_bf16(
                        false, af[mi], false, bfg[nj], (short)0, acc[mi][nj], false, false);
        }
        __syncthreads();   // all waves done reading buf before it is refilled
    }

    // epilogue: C layout -> global (row r / r+8 split by lane half)
#pragma unroll
    for (int nj = 0; nj < 2; ++nj) {
        const int gn = n0 + wn * 32 + nj * 16 + l15;
        const float bv = bias ? bias[gn] : 0.f;
#pragma unroll
        for (int mi = 0; mi < 4; ++mi) {
#pragma unroll
            for (int r = 0; r < 8; ++r) {
                const int gm = m0 + wm * 64 + mi * 16 + lhalf * 8 + r;
                const float v = acc[mi][nj][r] + bv;
                if (outf) outf[(size_t)gm * Ndim + gn] = v;
                else      outb[(size_t)gm * Ndim + gn] = (__bf16)v;
            }
        }
    }
}

// ---------------------------------------------------------------------------
// Flash attention over one dilation phase-batch / head.
// grid = (L/128, H, B*DIL); 256 thr = 8 waves; wave owns 16 queries.
// Q/K/V stored in original sequence order [B*S][D]; dilated gather is a
// strided row index: row = b*S + l*DIL + phase. K/V chunks double-buffered.
// ---------------------------------------------------------------------------
__global__ __launch_bounds__(256) void attn_wmma(
    const __bf16* __restrict__ Q, const __bf16* __restrict__ K,
    const __bf16* __restrict__ V, __bf16* __restrict__ ctx) {
    constexpr int Dm = 1024, Ss = 4096, Lq = 1024, DH = 64;
    constexpr int PAD = 8, KST = DH + PAD;             // 144B stride
    __shared__ __bf16 Ks[2][64][KST];                  // key-major (for Q*K^T)
    __shared__ __bf16 Vt[2][64][KST];                  // dh-major  (for P*V)
    __shared__ __bf16 Ps[8][16][KST];                  // per-wave probs staging

    const int tid = threadIdx.x;
    const int wave = tid >> 5, lane = tid & 31;
    const int lhalf = lane >> 4, l15 = lane & 15;
    const int bd = blockIdx.z, b = bd >> 2, phase = bd & 3;
    const int h = blockIdx.y;
    const int lq0 = blockIdx.x * 128 + wave * 16;

    // Q fragments (register-resident all kernel): 16 x 64
    const size_t qrow = (size_t)b * Ss + (size_t)(lq0 + l15) * 4 + phase;
    const __bf16* qp = Q + qrow * Dm + h * DH;
    v16bf qf[2];
#pragma unroll
    for (int kc = 0; kc < 2; ++kc) {
        const int base = kc * 32 + lhalf * 8;
        qf[kc] = load_frag2x16B(qp + base, qp + base + 16);
    }

    v8f cacc[4];
#pragma unroll
    for (int g = 0; g < 4; ++g) cacc[g] = zero8();
    float mrow[8], lrow[8];
#pragma unroll
    for (int r = 0; r < 8; ++r) { mrow[r] = -1e30f; lrow[r] = 0.f; }

    const int key = tid >> 2, dhoff = (tid & 3) * 16;
    auto issue_chunk = [&](int kk, int buf) {
        const size_t krow = (size_t)b * Ss + (size_t)(kk + key) * 4 + phase;
        const __bf16* kp = K + krow * Dm + h * DH + dhoff;
#if HAVE_ASYNC_LDS
        __builtin_amdgcn_global_load_async_to_lds_b128(
            to_as1(kp), to_as3(&Ks[buf][key][dhoff]), 0, 0);
        __builtin_amdgcn_global_load_async_to_lds_b128(
            to_as1(kp + 8), to_as3(&Ks[buf][key][dhoff + 8]), 0, 0);
#else
        uint4 k0 = *reinterpret_cast<const uint4*>(kp);
        uint4 k1 = *reinterpret_cast<const uint4*>(kp + 8);
        *reinterpret_cast<uint4*>(&Ks[buf][key][dhoff])     = k0;
        *reinterpret_cast<uint4*>(&Ks[buf][key][dhoff + 8]) = k1;
#endif
        const __bf16* vp = V + krow * Dm + h * DH + dhoff;
        FragU uv;
        uv.q[0] = *reinterpret_cast<const uint4*>(vp);
        uv.q[1] = *reinterpret_cast<const uint4*>(vp + 8);
#pragma unroll
        for (int i = 0; i < 16; ++i) Vt[buf][dhoff + i][key] = uv.e[i];  // transpose
    };

    constexpr int NCHUNK = Lq / 64;   // 16
    issue_chunk(0, 0);
    for (int c = 0; c < NCHUNK; ++c) {
        const int buf = c & 1;
        if (c + 1 < NCHUNK) {
            issue_chunk((c + 1) * 64, buf ^ 1);
#if HAVE_ASYNC_LDS
            __builtin_amdgcn_s_wait_asynccnt(2);  // newest 2 = next chunk's K
#endif
        } else {
#if HAVE_ASYNC_LDS
            __builtin_amdgcn_s_wait_asynccnt(0);
#endif
        }
        __syncthreads();

        // scores S = Q K^T / sqrt(dh): 4 key-groups x (k=64 -> 2 wmma)
        v8f sacc[4];
#pragma unroll
        for (int j = 0; j < 4; ++j) sacc[j] = zero8();
#pragma unroll
        for (int kc = 0; kc < 2; ++kc) {
            const int kb = kc * 32 + lhalf * 8;
#pragma unroll
            for (int j = 0; j < 4; ++j) {
                const int keyn = j * 16 + l15;
                v16bf kf = load_frag2x16B(&Ks[buf][keyn][kb], &Ks[buf][keyn][kb + 16]);
                sacc[j] = __builtin_amdgcn_wmma_f32_16x16x32_bf16(
                    false, qf[kc], false, kf, (short)0, sacc[j], false, false);
            }
        }

        // online softmax (rows live in one 16-lane half at fixed vgpr index r)
        float cf[8];
#pragma unroll
        for (int r = 0; r < 8; ++r) {
            float mx = -1e30f;
#pragma unroll
            for (int j = 0; j < 4; ++j) {
                const float s = sacc[j][r] * 0.125f;   // 1/sqrt(64)
                sacc[j][r] = s;
                mx = fmaxf(mx, s);
            }
#pragma unroll
            for (int off = 1; off < 16; off <<= 1) mx = fmaxf(mx, __shfl_xor(mx, off, 32));
            const float mn = fmaxf(mrow[r], mx);
            cf[r] = __expf(mrow[r] - mn);
            mrow[r] = mn;
            float rs = 0.f;
#pragma unroll
            for (int j = 0; j < 4; ++j) {
                const float p = __expf(sacc[j][r] - mn);
                sacc[j][r] = p;
                rs += p;
            }
#pragma unroll
            for (int off = 1; off < 16; off <<= 1) rs += __shfl_xor(rs, off, 32);
            lrow[r] = lrow[r] * cf[r] + rs;
        }
#pragma unroll
        for (int g = 0; g < 4; ++g)
#pragma unroll
            for (int r = 0; r < 8; ++r) cacc[g][r] *= cf[r];

        // probs: C-layout -> LDS row-major -> A-layout fragments
#pragma unroll
        for (int j = 0; j < 4; ++j)
#pragma unroll
            for (int r = 0; r < 8; ++r)
                Ps[wave][lhalf * 8 + r][j * 16 + l15] = (__bf16)sacc[j][r];
        __builtin_amdgcn_wave_barrier();   // wave-local LDS RAW; DS in-order per wave

        // ctx += P (16x64) * V (64x64)
#pragma unroll
        for (int kc = 0; kc < 2; ++kc) {
            const int ks2 = kc * 32 + lhalf * 8;
            v16bf pf = load_frag2x16B(&Ps[wave][l15][ks2], &Ps[wave][l15][ks2 + 16]);
            const int kb2 = kc * 32 + lhalf * 8;
#pragma unroll
            for (int g = 0; g < 4; ++g) {
                const int col = g * 16 + l15;
                v16bf vf = load_frag2x16B(&Vt[buf][col][kb2], &Vt[buf][col][kb2 + 16]);
                cacc[g] = __builtin_amdgcn_wmma_f32_16x16x32_bf16(
                    false, pf, false, vf, (short)0, cacc[g], false, false);
            }
        }
        __syncthreads();   // all waves done reading buf before it is refilled
    }

    // normalize and scatter ctx back into original sequence order (bf16)
#pragma unroll
    for (int r = 0; r < 8; ++r) {
        const int q = lq0 + lhalf * 8 + r;
        const size_t orow = (size_t)b * Ss + (size_t)q * 4 + phase;
        const float inv = 1.f / lrow[r];
#pragma unroll
        for (int g = 0; g < 4; ++g)
            ctx[orow * Dm + h * DH + g * 16 + l15] = (__bf16)(cacc[g][r] * inv);
    }
}

// ---------------------------------------------------------------------------
// final = LayerNorm(atted + x) * gamma + beta   (one row per block)
// ---------------------------------------------------------------------------
__global__ __launch_bounds__(256) void resid_ln(
    const float* __restrict__ atted, const float* __restrict__ x,
    const float* __restrict__ gamma, const float* __restrict__ beta,
    float* __restrict__ out) {
    constexpr int Dm = 1024;
    const int row = blockIdx.x, t = threadIdx.x;
    const float* ap = atted + (size_t)row * Dm;
    const float* xp = x + (size_t)row * Dm;
    float v[4], s = 0.f, ss = 0.f;
#pragma unroll
    for (int i = 0; i < 4; ++i) {
        const int c = t + i * 256;
        const float z = ap[c] + xp[c];
        v[i] = z; s += z; ss += z * z;
    }
#pragma unroll
    for (int off = 16; off > 0; off >>= 1) {
        s += __shfl_xor(s, off, 32);
        ss += __shfl_xor(ss, off, 32);
    }
    __shared__ float rs[8], rss[8];
    const int wave = t >> 5, lane = t & 31;
    if (lane == 0) { rs[wave] = s; rss[wave] = ss; }
    __syncthreads();
    float st = 0.f, sst = 0.f;
#pragma unroll
    for (int i = 0; i < 8; ++i) { st += rs[i]; sst += rss[i]; }
    const float mu = st * (1.f / Dm);
    const float var = sst * (1.f / Dm) - mu * mu;
    const float inv = rsqrtf(var + 1e-5f);
#pragma unroll
    for (int i = 0; i < 4; ++i) {
        const int c = t + i * 256;
        out[(size_t)row * Dm + c] = (v[i] - mu) * inv * gamma[c] + beta[c];
    }
}

// ---------------------------------------------------------------------------
extern "C" void kernel_launch(void* const* d_in, const int* in_sizes, int n_in,
                              void* d_out, int out_size, void* d_ws, size_t ws_size,
                              hipStream_t stream) {
    const float* x     = (const float*)d_in[0];
    const float* Wq    = (const float*)d_in[1];
    const float* bq    = (const float*)d_in[2];
    const float* Wk    = (const float*)d_in[3];
    const float* bk    = (const float*)d_in[4];
    const float* Wv    = (const float*)d_in[5];
    const float* bv    = (const float*)d_in[6];
    const float* Wf    = (const float*)d_in[7];
    const float* bfb   = (const float*)d_in[8];
    const float* gamma = (const float*)d_in[9];
    const float* beta  = (const float*)d_in[10];

    constexpr size_t MD = (size_t)8192 * 1024;   // B*S x D
    constexpr size_t DD = (size_t)1024 * 1024;   // D x D

    char* w = (char*)d_ws;
    __bf16* xb   = (__bf16*)w; w += MD * 2;
    __bf16* wqb  = (__bf16*)w; w += DD * 2;
    __bf16* wkb  = (__bf16*)w; w += DD * 2;
    __bf16* wvb  = (__bf16*)w; w += DD * 2;
    __bf16* wfb  = (__bf16*)w; w += DD * 2;
    __bf16* Qb   = (__bf16*)w; w += MD * 2;
    __bf16* Kb   = (__bf16*)w; w += MD * 2;
    __bf16* Vb   = (__bf16*)w; w += MD * 2;
    __bf16* ctxb = (__bf16*)w; w += MD * 2;      // ~88 MB total

    float* atted  = (float*)d_out + MD;          // tuple output #2
    float* finalp = (float*)d_out;               // tuple output #1

    // bf16 staging
    cvt_f32_bf16<<<(int)(MD / 4 / 256), 256, 0, stream>>>(x, xb, (int)(MD / 4));
    cvt_f32_bf16<<<(int)(DD / 4 / 256), 256, 0, stream>>>(Wq, wqb, (int)(DD / 4));
    cvt_f32_bf16<<<(int)(DD / 4 / 256), 256, 0, stream>>>(Wk, wkb, (int)(DD / 4));
    cvt_f32_bf16<<<(int)(DD / 4 / 256), 256, 0, stream>>>(Wv, wvb, (int)(DD / 4));
    cvt_f32_bf16<<<(int)(DD / 4 / 256), 256, 0, stream>>>(Wf, wfb, (int)(DD / 4));

    // QKV projections
    dim3 ggrid(64, 8);
    gemm_bf16_wmma<<<ggrid, 256, 0, stream>>>(xb, wqb, bq, Qb, nullptr, 8192, 1024, 1024);
    gemm_bf16_wmma<<<ggrid, 256, 0, stream>>>(xb, wkb, bk, Kb, nullptr, 8192, 1024, 1024);
    gemm_bf16_wmma<<<ggrid, 256, 0, stream>>>(xb, wvb, bv, Vb, nullptr, 8192, 1024, 1024);

    // dilated flash attention
    attn_wmma<<<dim3(8, 16, 8), 256, 0, stream>>>(Qb, Kb, Vb, ctxb);

    // output projection (f32) + bias
    gemm_bf16_wmma<<<ggrid, 256, 0, stream>>>(ctxb, wfb, bfb, nullptr, atted, 8192, 1024, 1024);

    // residual + LayerNorm
    resid_ln<<<8192, 256, 0, stream>>>(atted, x, gamma, beta, finalp);
}
